// TLForwardModel_86088324481536
// MI455X (gfx1250) — matrix operand order repaired and checked by reading, not compiled, not run on hardware
//
#include <hip/hip_runtime.h>

typedef __attribute__((ext_vector_type(16))) _Float16 v16h;
typedef __attribute__((ext_vector_type(8)))  _Float16 v8h;
typedef __attribute__((ext_vector_type(4)))  _Float16 v4h;
typedef __attribute__((ext_vector_type(8)))  float    v8f;
typedef __attribute__((ext_vector_type(4)))  float    v4f;
typedef __attribute__((ext_vector_type(4)))  unsigned u32x4;
typedef __attribute__((ext_vector_type(8)))  unsigned u32x8;

#define BATCH   4096
#define N_OBJ   64
#define KDIM    256   // GRID*GRID (x,y contraction)
#define NDIM    256   // GRID*GRID (p,q output)
#define N_ACT   16
#define KCHUNK  64
#define NCHUNKS (KDIM / KCHUNK)     // 4
#define LDS_STRIDE 72 // halves: 64 data + 16B pad -> 144B rows (16B aligned, bank spread)

// Low 32 bits of a generic pointer into LDS are the LDS byte offset
__device__ __forceinline__ unsigned lds_off(const void* p) {
    return (unsigned)(unsigned long long)p;
}

// One-shot: WT[a][n][k] = (f16) W[a][k][n]   (2 MB into workspace; W stays hot in L2)
__global__ void wt_transpose_kernel(const float* __restrict__ W, _Float16* __restrict__ WT) {
    int id = blockIdx.x * blockDim.x + threadIdx.x;      // 16*256*256 = 1M threads
    int k  = id & 255;
    int n  = (id >> 8) & 255;
    int a  = id >> 16;
    WT[(size_t)a * 65536 + (size_t)n * 256 + k] =
        (_Float16)W[(size_t)a * 65536 + (size_t)k * 256 + n];
}

__launch_bounds__(256)
__global__ void tl_forward_kernel(const float* __restrict__ state,
                                  const int*   __restrict__ action,
                                  const _Float16* __restrict__ WT,
                                  float* __restrict__ out) {
    // Double-buffered staging: 2*(9216 + 36864) = 92160 B of LDS (gfx1250: 320 KB/WGP)
    __shared__ _Float16 As[2][N_OBJ * LDS_STRIDE];
    __shared__ _Float16 Bs[2][NDIM * LDS_STRIDE];

    const int b    = blockIdx.x;
    const int t    = threadIdx.x;
    const int lane = t & 31;           // wave32
    const int wave = t >> 5;           // 0..7
    const int mt     = wave >> 1;      // M-tile 0..3
    const int ntile0 = (wave & 1) * 8; // 8 N-tiles per wave
    const int col  = lane & 15;
    const int hh   = lane >> 4;        // half-wave select

    const int a = action[b];
    const float*    sb = state + (size_t)b * (N_OBJ * KDIM);
    const _Float16* wa = WT    + (size_t)a * (NDIM * KDIM);

    // ---- TDM: DMA a 256x64 f16 tile of W^T into LDS with per-row padding ----
    // D# group0: count=1 | lds_addr | global_addr[56:0] | type=2
    // D# group1: data_size=2B, pad_enable, pad_interval=4 (32 DW = one 128B row),
    //            pad_amount=3 (4 DW = 16B)  -> LDS row stride 144B == LDS_STRIDE
    //            tensor_dim0=256, tensor_dim1=256, tile_dim0=64, tile_dim1=256,
    //            tensor_dim0_stride=256
    auto stage_B_tdm = [&](int buf, int kc) {
        unsigned long long ga = (unsigned long long)(const void*)(wa + kc);
        u32x4 g0;
        g0[0] = 1u;                                   // count=1, user mode
        g0[1] = lds_off(&Bs[buf][0]);                 // lds_addr
        g0[2] = (unsigned)ga;                         // global_addr[31:0]
        g0[3] = (unsigned)((ga >> 32) & 0x01ffffffu)  // global_addr[56:32]
              | (2u << 30);                           // type = 2 ("image")
        u32x8 g1;
        g1[0] = (1u << 16)                            // data_size = 1 -> 2 bytes
              | (1u << 20)                            // pad_enable
              | (4u << 22)                            // pad_interval: 2^(4+1)=32 DW
              | (3u << 25);                           // pad_amount: 4 DW (16B)
        g1[1] = (KDIM & 0xffffu) << 16;               // tensor_dim0[15:0] = 256
        g1[2] = ((unsigned)KDIM >> 16)                // tensor_dim0[31:16]
              | ((NDIM & 0xffffu) << 16);             // tensor_dim1[15:0] = 256
        g1[3] = ((unsigned)NDIM >> 16)                // tensor_dim1[31:16]
              | ((unsigned)KCHUNK << 16);             // tile_dim0 = 64
        g1[4] = (unsigned)NDIM;                       // tile_dim1=256, tile_dim2=0
        g1[5] = (unsigned)KDIM;                       // tensor_dim0_stride lo32 = 256
        g1[6] = 0u;                                   // stride hi16 | dim1_stride lo16
        g1[7] = 0u;                                   // dim1_stride hi32
        u32x4 gz = {0u, 0u, 0u, 0u};                  // groups 2/3: tile_dim3/4 = 0
        asm volatile("tensor_load_to_lds %0, %1, %2, %3"
                     :: "s"(g0), "s"(g1), "s"(gz), "s"(gz) : "memory");
    };

    // ---- VGPR stage of a 64x64 f32 chunk of state, converted to f16 ----
    auto stage_A = [&](int buf, int kc) {
#pragma unroll
        for (int i = 0; i < 4; ++i) {          // 1024 float4 / 256 threads
            int idx = t + 256 * i;
            int row = idx >> 4;
            int c4  = (idx & 15) << 2;
            v4f v = *(const v4f*)(sb + row * KDIM + kc + c4);
            v4h hv;
            hv[0] = (_Float16)v[0]; hv[1] = (_Float16)v[1];
            hv[2] = (_Float16)v[2]; hv[3] = (_Float16)v[3];
            *(v4h*)(&As[buf][row * LDS_STRIDE + c4]) = hv;
        }
    };

    v8f acc[8] = {};

    // Prologue: fill buffer 0 (one TDM per workgroup, issued by wave 0)
    if (wave == 0) stage_B_tdm(0, 0);
    stage_A(0, 0);
    if (wave == 0) __builtin_amdgcn_s_wait_tensorcnt(0);
    __syncthreads();

#pragma unroll
    for (int c = 0; c < NCHUNKS; ++c) {
        const int cur = c & 1;
        const int nxt = cur ^ 1;

        // Kick off next chunk's staging before computing (overlaps with WMMAs)
        if (c + 1 < NCHUNKS) {
            if (wave == 0) stage_B_tdm(nxt, (c + 1) * KCHUNK);
            stage_A(nxt, (c + 1) * KCHUNK);
        }

#pragma unroll
        for (int ks = 0; ks < 2; ++ks) {   // two K=32 WMMA steps per chunk
            // A fragment: 16-bit A 16x32 layout (lane=M; K: 8h..8h+7 then 16+8h..)
            const _Float16* arow = &As[cur][(mt * 16 + col) * LDS_STRIDE + ks * 32];
            v8h alo = *(const v8h*)(arow + 8 * hh);
            v8h ahi = *(const v8h*)(arow + 16 + 8 * hh);
            v16h af = __builtin_shufflevector(alo, ahi,
                        0,1,2,3,4,5,6,7,8,9,10,11,12,13,14,15);

            // Issue ALL 8 B-fragment loads first (16 ds_load_b128 in flight),
            // then drain with WMMAs -> staggered s_wait_dscnt instead of 0x0 stalls.
            v16h bf[8];
#pragma unroll
            for (int nt = 0; nt < 8; ++nt) {
                const _Float16* brow =
                    &Bs[cur][((ntile0 + nt) * 16 + col) * LDS_STRIDE + ks * 32 + 8 * hh];
                v8h blo = *(const v8h*)brow;
                v8h bhi = *(const v8h*)(brow + 16);
                bf[nt] = __builtin_shufflevector(blo, bhi,
                            0,1,2,3,4,5,6,7,8,9,10,11,12,13,14,15);
            }
#pragma unroll
            for (int nt = 0; nt < 8; ++nt) {
                acc[nt] = __builtin_amdgcn_wmma_f32_16x16x32_f16(
                    /*neg_a=*/false, af, /*neg_b=*/false, bf[nt],
                    /*c_mod=*/(short)0, acc[nt],
                    /*reuse_a=*/false, /*reuse_b=*/false);
            }
        }

        // Wave 0: my TDM into 'nxt' done; barrier: all waves see it, and all
        // waves are finished reading 'cur' fragments.
        if (wave == 0) __builtin_amdgcn_s_wait_tensorcnt(0);
        __syncthreads();
    }

    // ---- epilogue: sigmoid + store (C/D layout: vgpr v -> M = v + 8*hh, N = col) ----
    float* ob = out + (size_t)b * (N_OBJ * NDIM);
#pragma unroll
    for (int nt = 0; nt < 8; ++nt) {
        int n0 = (ntile0 + nt) * 16 + col;
#pragma unroll
        for (int v = 0; v < 8; ++v) {
            int m = mt * 16 + v + 8 * hh;
            float s = acc[nt][v];
            ob[m * NDIM + n0] = 1.0f / (1.0f + __expf(-s));
        }
    }
}

extern "C" void kernel_launch(void* const* d_in, const int* in_sizes, int n_in,
                              void* d_out, int out_size, void* d_ws, size_t ws_size,
                              hipStream_t stream) {
    const float* state  = (const float*)d_in[0];
    const int*   action = (const int*)  d_in[1];
    const float* W      = (const float*)d_in[2];
    float*       out    = (float*)d_out;
    _Float16*    WT     = (_Float16*)d_ws;   // needs 16*256*256*2 = 2 MB scratch

    // 1) transpose + f16-convert W into workspace (k-major -> n-major)
    wt_transpose_kernel<<<(N_ACT * KDIM * NDIM) / 256, 256, 0, stream>>>(W, WT);
    // 2) batched WMMA GEMM + sigmoid, one block per batch element
    tl_forward_kernel<<<BATCH, 256, 0, stream>>>(state, action, WT, out);
}
